// WeightOnlyInt4Linear_51393578664546
// MI455X (gfx1250) — compile-verified, matrix operand order checked by reading, not run on hardware
//
#include <hip/hip_runtime.h>
#include <hip/hip_bf16.h>

typedef __attribute__((ext_vector_type(16))) _Float16 v16h;
typedef __attribute__((ext_vector_type(8)))  _Float16 v8h;
typedef __attribute__((ext_vector_type(4)))  _Float16 v4h;
typedef __attribute__((ext_vector_type(8)))  float    v8f;
typedef __attribute__((ext_vector_type(4)))  float    v4f;
typedef __attribute__((ext_vector_type(2)))  float    v2f;
typedef __attribute__((ext_vector_type(4)))  int      v4i;

#define M_TOT 512
#define N_TOT 11008
#define K_TOT 4096
#define BM 128
#define BN 128
#define BK 64
#define LDSS 72   // padded row stride in halfs (144 B, 16B-aligned, breaks bank conflicts)

union FragU { v16h v; v8h h[2]; };

__global__ __launch_bounds__(256)
void wq4_wmma_gemm(const float* __restrict__ x,
                   const int*   __restrict__ wq,
                   const float* __restrict__ saz,
                   float*       __restrict__ out)
{
    __shared__ __align__(16) _Float16 As[BM * LDSS];
    __shared__ __align__(16) _Float16 Bs[BN * LDSS];

    const int tid  = threadIdx.x;
    const int n0   = blockIdx.x * BN;     // 0..85 * 128
    const int m0   = blockIdx.y * BM;     // 0..3  * 128

    const int lane = tid & 31;            // wave32
    const int wv   = tid >> 5;            // 8 waves
    const int l16  = lane & 15;
    const int hi   = lane >> 4;           // 0/1 lane-half
    const int wm   = wv >> 1;             // 0..3 -> 32 M rows each
    const int wn   = wv & 1;              // 0..1 -> 64 N cols each

    // Loader decomposition: 16 threads per row, 4 elements each (256B segments / 16 lanes)
    const int lk = tid & 15;              // k chunk within row (4 elems)
    const int lr = tid >> 4;              // base row, step 16, 8 iters -> 128 rows

    v8f acc[2][4];
#pragma unroll
    for (int i = 0; i < 2; ++i)
#pragma unroll
        for (int j = 0; j < 4; ++j)
#pragma unroll
            for (int r = 0; r < 8; ++r) acc[i][j][r] = 0.0f;

    for (int kt = 0; kt < K_TOT / BK; ++kt) {
        const int k0 = kt * BK;
        const int g  = k0 >> 7;           // GS=128, BK=64 -> group constant per step

        // Prefetch next K-step's weight line (global_prefetch_b8)
        if (kt + 1 < K_TOT / BK) {
            __builtin_prefetch(
                (const void*)(wq + (long)(n0 + lr) * K_TOT + (k0 + BK) + lk * 4), 0, 1);
        }

        // ---- A tile: x[m0+row][k0 + lk*4 ..] fp32 -> f16 -> LDS ----
#pragma unroll
        for (int i = 0; i < 8; ++i) {
            const int row = lr + i * 16;
            v4f a = *(const v4f*)(x + (long)(m0 + row) * K_TOT + k0 + lk * 4);
            v4h h;
            h.x = (_Float16)a.x; h.y = (_Float16)a.y;
            h.z = (_Float16)a.z; h.w = (_Float16)a.w;
            *(v4h*)(As + row * LDSS + lk * 4) = h;
        }

        // ---- B tile: dequant wq[n0+row][k0 + lk*4 ..] -> f16 -> LDS ----
#pragma unroll
        for (int i = 0; i < 8; ++i) {
            const int row = lr + i * 16;
            v4i q  = *(const v4i*)(wq + (long)(n0 + row) * K_TOT + k0 + lk * 4);
            v2f sz = *(const v2f*)(saz + ((long)g * N_TOT + (n0 + row)) * 2);
            const float s = sz.x, z = sz.y;
            v4h h;
            h.x = (_Float16)((float)(q.x - 8) * s + z);
            h.y = (_Float16)((float)(q.y - 8) * s + z);
            h.z = (_Float16)((float)(q.z - 8) * s + z);
            h.w = (_Float16)((float)(q.w - 8) * s + z);
            *(v4h*)(Bs + row * LDSS + lk * 4) = h;
        }

        __syncthreads();

        // ---- Compute: 2 K-substeps of 32, 8 WMMAs each per wave ----
#pragma unroll
        for (int ks = 0; ks < 2; ++ks) {
            const int kb = ks * 32 + hi * 8;  // lane-half K offset (ISA 16-bit A/B layout)

            FragU af[2];
#pragma unroll
            for (int im = 0; im < 2; ++im) {
                const _Float16* p = As + (wm * 32 + im * 16 + l16) * LDSS + kb;
                af[im].h[0] = *(const v8h*)(p);
                af[im].h[1] = *(const v8h*)(p + 16);
            }

            FragU bf[4];
#pragma unroll
            for (int in = 0; in < 4; ++in) {
                const _Float16* p = Bs + (wn * 64 + in * 16 + l16) * LDSS + kb;
                bf[in].h[0] = *(const v8h*)(p);
                bf[in].h[1] = *(const v8h*)(p + 16);
            }

#pragma unroll
            for (int im = 0; im < 2; ++im)
#pragma unroll
                for (int in = 0; in < 4; ++in) {
                    acc[im][in] = __builtin_amdgcn_wmma_f32_16x16x32_f16(
                        false, af[im].v, false, bf[in].v,
                        (short)0, acc[im][in], false, false);
                }
        }

        __syncthreads();
    }

    // ---- Store: C layout -> lane l: N = l%16; VGPR r: M = r + 8*(l>=16) ----
#pragma unroll
    for (int im = 0; im < 2; ++im) {
#pragma unroll
        for (int in = 0; in < 4; ++in) {
            const int n = n0 + wn * 64 + in * 16 + l16;
#pragma unroll
            for (int r = 0; r < 8; ++r) {
                const int m = m0 + wm * 32 + im * 16 + r + 8 * hi;
                out[(long)m * N_TOT + n] = acc[im][in][r];
            }
        }
    }
}

extern "C" void kernel_launch(void* const* d_in, const int* in_sizes, int n_in,
                              void* d_out, int out_size, void* d_ws, size_t ws_size,
                              hipStream_t stream) {
    const float* x   = (const float*)d_in[0];
    const int*   wq  = (const int*)d_in[1];
    const float* saz = (const float*)d_in[2];
    float*       out = (float*)d_out;

    dim3 grid(N_TOT / BN, M_TOT / BM);   // (86, 4)
    dim3 block(256);
    wq4_wmma_gemm<<<grid, block, 0, stream>>>(x, wq, saz, out);
}
